// SingleLayerMambaPGF_24945170055931
// MI455X (gfx1250) — matrix operand order repaired and checked by reading, not compiled, not run on hardware
//
#include <hip/hip_runtime.h>

#define L_SEQ   4096
#define DMODEL  1024
#define DSTATE  16
#define NCHUNK  32
#define LCHUNK  128   // L_SEQ / NCHUNK

typedef __attribute__((ext_vector_type(16))) __bf16 v16bf;
typedef __attribute__((ext_vector_type(8)))  float  v8f;
typedef __attribute__((ext_vector_type(4)))  unsigned v4u;
typedef __attribute__((ext_vector_type(8)))  int    v8i;
typedef __attribute__((ext_vector_type(4)))  int    v4i;

union FragBF {
  uint4 q[2];
  v16bf v;
};

// reuse_a must be an immediate -> template parameter.
template <bool RA>
__device__ __forceinline__ v8f wmma_bf16(v16bf a, v16bf b, v8f c) {
  return __builtin_amdgcn_wmma_f32_16x16x32_bf16(false, a, false, b,
                                                 (short)0, c, RA, false);
}

__device__ __forceinline__ unsigned short f2bf(float f) {
  unsigned u = __float_as_uint(f);
  u += 0x7FFFu + ((u >> 16) & 1u);   // round-to-nearest-even
  return (unsigned short)(u >> 16);
}

// ---------------------------------------------------------------- convert
__global__ void cvt_bf16_kernel(const float* __restrict__ src,
                                unsigned short* __restrict__ dst, int n) {
  int i = blockIdx.x * blockDim.x + threadIdx.x;
  if (i < n) dst[i] = f2bf(src[i]);
}

// ------------------------------------------------------------- TDM helper
// 2D tile load: `rows` rows of 64 bf16 elements, source row stride 1024
// elements, destination LDS padded +4 dwords per 32-dword row
// (pitch = 144 B -> 16B aligned, 36-bank row stride: conflict-free).
__device__ __forceinline__ void tdm_load_tile(unsigned lds_addr,
                                              const void* gp,
                                              unsigned rows) {
  unsigned long long ga = (unsigned long long)(size_t)gp;
  v4u g0;
  g0[0] = 1u;                                        // count=1 (valid), user mode
  g0[1] = lds_addr;                                  // LDS byte address
  g0[2] = (unsigned)ga;                              // global_addr[31:0]
  g0[3] = (unsigned)((ga >> 32) & 0x1FFFFFFu)        // global_addr[56:32]
        | 0x80000000u;                               // type=2 ("image")
  v8i g1;
  g1[0] = (int)((1u << 16)                           // data_size = 2 bytes
              | (1u << 20)                           // pad_enable
              | (4u << 22)                           // pad_interval: every 32 dwords
              | (3u << 25));                         // pad_amount: 4 dwords
  g1[1] = (int)(64u << 16);                          // tensor_dim0 = 64 (lo16)
  g1[2] = (int)((rows & 0xFFFFu) << 16);             // tensor_dim1 = rows (lo16)
  g1[3] = (int)(64u << 16);                          // tile_dim0 = 64
  g1[4] = (int)(rows & 0xFFFFu);                     // tile_dim1 = rows, tile_dim2 = 0
  g1[5] = 1024;                                      // tensor_dim0_stride (lo32)
  g1[6] = 0;
  g1[7] = 0;
  v4i gz = (v4i){0, 0, 0, 0};
#if defined(__clang_major__) && __clang_major__ >= 23
  __builtin_amdgcn_tensor_load_to_lds(g0, g1, gz, gz, (v8i){0,0,0,0,0,0,0,0}, 0);
#else
  __builtin_amdgcn_tensor_load_to_lds(g0, g1, gz, gz, 0);
#endif
}

// ------------------------------------------------- dt = softplus(u @ Wt + b)
// Block tile 64(M) x 128(N), K staged 64 deep via TDM double-buffer in LDS.
// 8 waves as 4(M) x 2(N); each wave computes 16x64 (4 WMMA accumulators).
// grid = (DMODEL/128, L_SEQ/64) = (8, 64), block = 256.
#define LDS_PITCH   72      // ushorts per row (144 B, incl. TDM pad)
#define A_STAGE_USH (64 * LDS_PITCH)            // 4608 ushorts (9216 B)
#define B_STAGE_USH (128 * LDS_PITCH)           // 9216 ushorts (18432 B)
#define STAGE_USH   (A_STAGE_USH + B_STAGE_USH) // 13824 ushorts (27648 B)
#define NSTAGE      (DMODEL / 64)               // 16

__global__ __launch_bounds__(256)
void gemm_dt_kernel(const unsigned short* __restrict__ u_bf,
                    const unsigned short* __restrict__ w_bf,
                    const float* __restrict__ dt_b,
                    float* __restrict__ dt_out) {
  __shared__ __align__(16) unsigned short lds_tile[2 * STAGE_USH];

  const int lane = threadIdx.x & 31;
  const int wave = threadIdx.x >> 5;
  const int lr   = lane & 15;
  const int half = lane >> 4;
  const int wm   = wave >> 1;   // 0..3 : M sub-tile
  const int wn   = wave & 1;    // 0..1 : N sub-tile

  const unsigned lds_base = (unsigned)(size_t)(void*)lds_tile;
  const unsigned short* gA = u_bf + (size_t)blockIdx.y * 64 * DMODEL;
  const unsigned short* gB = w_bf + (size_t)blockIdx.x * 128 * DMODEL;

  // Per-wave LDS read bases (ushort units).
  const unsigned short* aRow = lds_tile + (wm * 16 + lr) * LDS_PITCH;
  const unsigned short* bRow = lds_tile + A_STAGE_USH + (wn * 64 + lr) * LDS_PITCH;

  v8f acc[4] = {};

  // Prologue: stage 0 into buffer 0.
  if (wave == 0) {
    tdm_load_tile(lds_base,                   gA, 64);
    tdm_load_tile(lds_base + 2 * A_STAGE_USH, gB, 128);
    __builtin_amdgcn_s_wait_tensorcnt(0);
  }
  __syncthreads();

  for (int s = 0; s < NSTAGE; ++s) {
    const int buf = s & 1;

    // Issue next stage into the other buffer while computing this one.
    if (wave == 0 && s + 1 < NSTAGE) {
      const unsigned dst = lds_base + (unsigned)((buf ^ 1) * (2 * STAGE_USH));
      tdm_load_tile(dst,                    gA + (s + 1) * 64, 64);
      tdm_load_tile(dst + 2 * A_STAGE_USH,  gB + (s + 1) * 64, 128);
    }

    const unsigned short* aS = aRow + buf * STAGE_USH;
    const unsigned short* bS = bRow + buf * STAGE_USH;

    // Hoist ALL fragment loads for this stage, then run 8 WMMAs
    // back-to-back (one s_wait_dscnt instead of eight).
    FragBF fa[2];
    FragBF fb[2][4];
#pragma unroll
    for (int ks2 = 0; ks2 < 2; ++ks2) {
      const int ks = ks2 * 32;
      fa[ks2].q[0] = *reinterpret_cast<const uint4*>(aS + ks + half * 16);
      fa[ks2].q[1] = *reinterpret_cast<const uint4*>(aS + ks + half * 16 + 8);
#pragma unroll
      for (int t = 0; t < 4; ++t) {
        const unsigned short* bp = bS + t * 16 * LDS_PITCH + ks + half * 16;
        fb[ks2][t].q[0] = *reinterpret_cast<const uint4*>(bp);
        fb[ks2][t].q[1] = *reinterpret_cast<const uint4*>(bp + 8);
      }
    }
#pragma unroll
    for (int ks2 = 0; ks2 < 2; ++ks2) {
      // reuse_a on WMMAs 1..3: same A fragment as the immediately preceding
      // identical instruction (ISA §7.12 RA hint). Immediates via template.
      acc[0] = wmma_bf16<false>(fa[ks2].v, fb[ks2][0].v, acc[0]);
      acc[1] = wmma_bf16<true >(fa[ks2].v, fb[ks2][1].v, acc[1]);
      acc[2] = wmma_bf16<true >(fa[ks2].v, fb[ks2][2].v, acc[2]);
      acc[3] = wmma_bf16<true >(fa[ks2].v, fb[ks2][3].v, acc[3]);
    }

    // Current stage consumed; make sure next stage's TDM finished before the
    // whole block flips buffers.
    if (wave == 0 && s + 1 < NSTAGE) __builtin_amdgcn_s_wait_tensorcnt(0);
    __syncthreads();
  }

  const int mbase = blockIdx.y * 64 + wm * 16;
  const int nbase = blockIdx.x * 128 + wn * 64;
#pragma unroll
  for (int t = 0; t < 4; ++t) {
    const int col  = nbase + t * 16 + lr;
    const float bias = dt_b[col];
#pragma unroll
    for (int r = 0; r < 8; ++r) {
      const int row = mbase + r + half * 8;
      float x  = acc[t][r] + bias;
      float sp = (x > 20.0f) ? x : log1pf(expf(x));   // softplus
      dt_out[row * DMODEL + col] = sp;
    }
  }
}

// ------------------------------------- B_vec / C_vec = u @ {B_w,C_w}^T + bias
// block = 64 (2 waves): wave0 -> B, wave1 -> C. grid = 256 M-tiles.
__global__ __launch_bounds__(64)
void gemm_bc_kernel(const unsigned short* __restrict__ u_bf,
                    const unsigned short* __restrict__ Bw_bf,
                    const unsigned short* __restrict__ Cw_bf,
                    const float* __restrict__ B_b,
                    const float* __restrict__ C_b,
                    float* __restrict__ Bv,
                    float* __restrict__ Cv) {
  const int lane = threadIdx.x & 31;
  const int wave = threadIdx.x >> 5;
  const int lr   = lane & 15;
  const int half = lane >> 4;
  const int mbase = blockIdx.x * 16;

  const unsigned short* wmat = wave ? Cw_bf : Bw_bf;
  const float* bias = wave ? C_b : B_b;
  float* out = wave ? Cv : Bv;

  const unsigned short* aRow = u_bf + (mbase + lr) * DMODEL;
  const unsigned short* bRow = wmat + lr * DMODEL;

  v8f acc = {};
  for (int k0 = 0; k0 < DMODEL; k0 += 32) {
    FragBF fa, fb;
    const int ka = k0 + half * 16;
    fa.q[0] = *reinterpret_cast<const uint4*>(aRow + ka);
    fa.q[1] = *reinterpret_cast<const uint4*>(aRow + ka + 8);
    const int kb = k0 + half * 16;
    fb.q[0] = *reinterpret_cast<const uint4*>(bRow + kb);
    fb.q[1] = *reinterpret_cast<const uint4*>(bRow + kb + 8);
    acc = wmma_bf16<false>(fa.v, fb.v, acc);
    if (k0 + 32 < DMODEL)
      __builtin_prefetch((const void*)(aRow + ka + 32), 0, 3);
  }
  const float bs = bias[lr];
#pragma unroll
  for (int r = 0; r < 8; ++r) {
    const int row = mbase + r + half * 8;
    out[row * DSTATE + lr] = acc[r] + bs;
  }
}

// ----------------------------------------- scan pass A: per-chunk (P, S)
__global__ __launch_bounds__(256)
void scan_passA_kernel(const float* __restrict__ dt,
                       const float* __restrict__ u,
                       const float* __restrict__ Bv,
                       const float* __restrict__ A_log,
                       float* __restrict__ P_arr,
                       float* __restrict__ S_arr) {
  const int n = threadIdx.x & 15;
  const int d = blockIdx.x * 16 + (threadIdx.x >> 4);
  const int c = blockIdx.y;
  const float A = -expf(A_log[n]);
  float P = 1.0f, S = 0.0f;
  const int l0 = c * LCHUNK;
  for (int l = l0; l < l0 + LCHUNK; ++l) {
    const float dtv = dt[l * DMODEL + d];
    const float z   = dtv * A;
    const float em  = expm1f(z);          // one transcendental
    const float dA  = em + 1.0f;          // exp(z)
    const float phi = em / (z + 1e-9f);   // phi_1(z)
    const float b   = phi * Bv[l * DSTATE + n] * u[l * DMODEL + d];
    P *= dA;
    S = fmaf(dA, S, b);
  }
  const int o = (c * DMODEL + d) * DSTATE + n;
  P_arr[o] = P;
  S_arr[o] = S;
}

// ------------------------------- scan pass B: scan over chunk boundaries
__global__ void scan_passB_kernel(const float* __restrict__ P_arr,
                                  const float* __restrict__ S_arr,
                                  float* __restrict__ Init) {
  const int idx = blockIdx.x * blockDim.x + threadIdx.x;  // d*16 + n
  float H = 0.0f;
  for (int c = 0; c < NCHUNK; ++c) {
    const int o = c * (DMODEL * DSTATE) + idx;
    Init[o] = H;                       // state entering chunk c
    H = fmaf(P_arr[o], H, S_arr[o]);
  }
}

// ---------------- scan pass C: replay with correct init + output reduce
__global__ __launch_bounds__(256)
void scan_passC_kernel(const float* __restrict__ dt,
                       const float* __restrict__ u,
                       const float* __restrict__ Bv,
                       const float* __restrict__ Cv,
                       const float* __restrict__ A_log,
                       const float* __restrict__ D_param,
                       const float* __restrict__ scale_p,
                       const float* __restrict__ Init,
                       float* __restrict__ y) {
  const int n = threadIdx.x & 15;
  const int d = blockIdx.x * 16 + (threadIdx.x >> 4);
  const int c = blockIdx.y;
  const float A  = -expf(A_log[n]);
  const float Dp = D_param[d];
  const float sc = scale_p[0];
  float h = Init[(c * DMODEL + d) * DSTATE + n];
  const int l0 = c * LCHUNK;
  for (int l = l0; l < l0 + LCHUNK; ++l) {
    const float dtv = dt[l * DMODEL + d];
    const float uv  = u[l * DMODEL + d];
    const float z   = dtv * A;
    const float em  = expm1f(z);
    const float dA  = em + 1.0f;
    const float phi = em / (z + 1e-9f);
    const float b   = phi * Bv[l * DSTATE + n] * uv;
    h = fmaf(dA, h, b);
    float yc = Cv[l * DSTATE + n] * h;
#pragma unroll
    for (int m = 8; m >= 1; m >>= 1) yc += __shfl_xor(yc, m, 16);  // reduce N
    if (n == 0) y[l * DMODEL + d] = fmaf(Dp, uv, yc) * sc;
  }
}

// ---------------------------------------------------------------- launch
extern "C" void kernel_launch(void* const* d_in, const int* in_sizes, int n_in,
                              void* d_out, int out_size, void* d_ws, size_t ws_size,
                              hipStream_t stream) {
  (void)in_sizes; (void)n_in; (void)out_size; (void)ws_size;
  const float* u      = (const float*)d_in[0];
  const float* A_log  = (const float*)d_in[1];
  const float* D_par  = (const float*)d_in[2];
  const float* scale  = (const float*)d_in[3];
  const float* dt_w   = (const float*)d_in[4];
  const float* dt_b   = (const float*)d_in[5];
  const float* B_w    = (const float*)d_in[6];
  const float* B_b    = (const float*)d_in[7];
  const float* C_w    = (const float*)d_in[8];
  const float* C_b    = (const float*)d_in[9];
  float* y = (float*)d_out;

  char* ws = (char*)d_ws;
  size_t off = 0;
  auto take = [&](size_t bytes) { char* p = ws + off; off = (off + bytes + 255) & ~size_t(255); return p; };
  unsigned short* u_bf  = (unsigned short*)take((size_t)L_SEQ * DMODEL * 2);
  unsigned short* w_bf  = (unsigned short*)take((size_t)DMODEL * DMODEL * 2);
  unsigned short* Bw_bf = (unsigned short*)take((size_t)DSTATE * DMODEL * 2);
  unsigned short* Cw_bf = (unsigned short*)take((size_t)DSTATE * DMODEL * 2);
  float* dt_ws = (float*)take((size_t)L_SEQ * DMODEL * 4);
  float* Bv    = (float*)take((size_t)L_SEQ * DSTATE * 4);
  float* Cv    = (float*)take((size_t)L_SEQ * DSTATE * 4);
  float* P_arr = (float*)take((size_t)NCHUNK * DMODEL * DSTATE * 4);
  float* S_arr = (float*)take((size_t)NCHUNK * DMODEL * DSTATE * 4);
  float* Initv = (float*)take((size_t)NCHUNK * DMODEL * DSTATE * 4);

  // 1) fp32 -> bf16 for WMMA operands
  {
    int n = L_SEQ * DMODEL;
    cvt_bf16_kernel<<<(n + 255) / 256, 256, 0, stream>>>(u, u_bf, n);
    n = DMODEL * DMODEL;
    cvt_bf16_kernel<<<(n + 255) / 256, 256, 0, stream>>>(dt_w, w_bf, n);
    n = DSTATE * DMODEL;
    cvt_bf16_kernel<<<(n + 255) / 256, 256, 0, stream>>>(B_w, Bw_bf, n);
    cvt_bf16_kernel<<<(n + 255) / 256, 256, 0, stream>>>(C_w, Cw_bf, n);
  }

  // 2) WMMA projections (dt GEMM is TDM->LDS staged, double-buffered)
  gemm_dt_kernel<<<dim3(DMODEL / 128, L_SEQ / 64), 256, 0, stream>>>(
      u_bf, w_bf, dt_b, dt_ws);
  gemm_bc_kernel<<<dim3(L_SEQ / 16), 64, 0, stream>>>(u_bf, Bw_bf, Cw_bf,
                                                      B_b, C_b, Bv, Cv);

  // 3) chunked associative scan + output
  scan_passA_kernel<<<dim3(DMODEL / 16, NCHUNK), 256, 0, stream>>>(
      dt_ws, u, Bv, A_log, P_arr, S_arr);
  scan_passB_kernel<<<dim3((DMODEL * DSTATE) / 256), 256, 0, stream>>>(
      P_arr, S_arr, Initv);
  scan_passC_kernel<<<dim3(DMODEL / 16, NCHUNK), 256, 0, stream>>>(
      dt_ws, u, Bv, Cv, A_log, D_par, scale, Initv, y);
}
